// Q2A_Function_blip_90915867722507
// MI455X (gfx1250) — matrix operand order repaired and checked by reading, not compiled, not run on hardware
//
#include <hip/hip_runtime.h>
#include <hip/hip_bf16.h>

// ---------------------------------------------------------------------------
// MI455X / gfx1250. All heavy matmuls via v_wmma_f32_16x16x32_bf16 (wave32,
// f32 accum). f32 operands are converted to bf16 while staging into LDS.
// All GEMM M/N dims are padded to multiples of 128 so staging and epilogue
// are guard-free straight-line vector code. bf16 A tiles are copied
// global->LDS with GLOBAL_LOAD_ASYNC_TO_LDS_B128 (ASYNCcnt).
// ---------------------------------------------------------------------------

typedef __attribute__((ext_vector_type(16))) __bf16 v16bf;
typedef __attribute__((ext_vector_type(8)))  float  v8f;

#if defined(__gfx1250__) && __has_builtin(__builtin_amdgcn_global_load_async_to_lds_b128)
#define HAVE_ASYNC_LDS 1
typedef int v4i_t __attribute__((vector_size(16)));
typedef __attribute__((address_space(1))) v4i_t* v4i_gptr;
typedef __attribute__((address_space(3))) v4i_t* v4i_lptr;
#else
#define HAVE_ASYNC_LDS 0
#endif

static __device__ __forceinline__ unsigned short f32_to_bf16(float f) {
    unsigned int u = __float_as_uint(f);
    unsigned int r = u + 0x7FFFu + ((u >> 16) & 1u);   // round-to-nearest-even
    return (unsigned short)(r >> 16);
}
static __device__ __forceinline__ float bf16_to_f32(unsigned short h) {
    return __uint_as_float(((unsigned int)h) << 16);
}
static __device__ __forceinline__ unsigned int pk_bf16(float lo, float hi) {
    return (unsigned int)f32_to_bf16(lo) | ((unsigned int)f32_to_bf16(hi) << 16);
}
static __device__ __forceinline__ void wait_asynccnt0() {
#if defined(__gfx1250__)
#if __has_builtin(__builtin_amdgcn_s_wait_asynccnt)
    __builtin_amdgcn_s_wait_asynccnt(0);
#else
    asm volatile("s_wait_asynccnt 0x0" ::: "memory");
#endif
#endif
}

// ---------------------------------------------------------------------------
// Tiled bf16 WMMA GEMM:  C(MxN) = act(A(MxK) @ B(KxN) + bias)
//   A: f32 or bf16 (A_BF16), row-major. M, N multiples of 128; K of 32.
//   B: f32 weights. If B_TRANS, B is (NxK) row-major and we compute A @ B^T.
//   Block: 256 threads = 8 waves -> 128x128 C tile (wave grid 4m x 2n, each
//   wave 32x64 = 2x4 fragments of 16x16). K tiled by 32 through LDS.
// ---------------------------------------------------------------------------
template<bool A_BF16, bool B_TRANS, bool RELU>
__global__ __launch_bounds__(256)
void gemm_wmma(const void* __restrict__ Av, const float* __restrict__ B,
               const float* __restrict__ bias,
               float* __restrict__ outF, unsigned short* __restrict__ outH,
               int M, int N, int K)
{
    __shared__ unsigned short lsA[128 * 32];  // [m][k] bf16
    __shared__ unsigned short lsB[128 * 32];  // [n][k] bf16 (transposed tile)

    const int tid  = threadIdx.x;
    const int lane = tid & 31;
    const int wave = tid >> 5;
    const int wm   = (wave & 3) * 32;   // wave M offset within tile
    const int wn   = (wave >> 2) * 64;  // wave N offset within tile
    const int bm   = blockIdx.y * 128;
    const int bn   = blockIdx.x * 128;

    const float*          Af = (const float*)Av;
    const unsigned short* Ah = (const unsigned short*)Av;

    v8f acc[2][4] = {};

    const int lrow = lane & 15;
    const int lhi  = lane >> 4;

    // staging coordinates: each thread owns half a row (16 elements)
    const int srow = tid >> 1;          // 0..127
    const int scol = (tid & 1) * 16;    // 0 or 16

    union Frag { unsigned int u[8]; v16bf v; };

    for (int k0 = 0; k0 < K; k0 += 32) {
        // ---- stage A tile (128x32) ----
        if (A_BF16) {
#if HAVE_ASYNC_LDS
            const unsigned short* g = &Ah[(size_t)(bm + srow) * K + k0 + scol];
            __builtin_amdgcn_global_load_async_to_lds_b128(
                (v4i_gptr)(g), (v4i_lptr)&lsA[srow * 32 + scol], 0, 0);
            __builtin_amdgcn_global_load_async_to_lds_b128(
                (v4i_gptr)(g + 8), (v4i_lptr)&lsA[srow * 32 + scol + 8], 0, 0);
#else
            const uint4* g = (const uint4*)&Ah[(size_t)(bm + srow) * K + k0 + scol];
            uint4* d = (uint4*)&lsA[srow * 32 + scol];
            d[0] = g[0];
            d[1] = g[1];
#endif
        } else {
            const float4* g = (const float4*)&Af[(size_t)(bm + srow) * K + k0 + scol];
            float4 a0 = g[0], a1 = g[1], a2 = g[2], a3 = g[3];
            uint4 o0, o1;
            o0.x = pk_bf16(a0.x, a0.y); o0.y = pk_bf16(a0.z, a0.w);
            o0.z = pk_bf16(a1.x, a1.y); o0.w = pk_bf16(a1.z, a1.w);
            o1.x = pk_bf16(a2.x, a2.y); o1.y = pk_bf16(a2.z, a2.w);
            o1.z = pk_bf16(a3.x, a3.y); o1.w = pk_bf16(a3.z, a3.w);
            uint4* d = (uint4*)&lsA[srow * 32 + scol];
            d[0] = o0;
            d[1] = o1;
        }
        // ---- stage B tile into [n][k] ----
        if (B_TRANS) {
            // B is (N x K) row-major: lsB[n][k] <- B[bn+n][k0+k], contiguous k
            const float4* g = (const float4*)&B[(size_t)(bn + srow) * K + k0 + scol];
            float4 b0 = g[0], b1 = g[1], b2 = g[2], b3 = g[3];
            uint4 o0, o1;
            o0.x = pk_bf16(b0.x, b0.y); o0.y = pk_bf16(b0.z, b0.w);
            o0.z = pk_bf16(b1.x, b1.y); o0.w = pk_bf16(b1.z, b1.w);
            o1.x = pk_bf16(b2.x, b2.y); o1.y = pk_bf16(b2.z, b2.w);
            o1.z = pk_bf16(b3.x, b3.y); o1.w = pk_bf16(b3.z, b3.w);
            uint4* d = (uint4*)&lsB[srow * 32 + scol];
            d[0] = o0;
            d[1] = o1;
        } else {
            // B is (K x N) row-major: vector-read 4 columns, scatter by n
            #pragma unroll
            for (int j = 0; j < 4; ++j) {
                int i4 = tid + 256 * j;       // 1024 float4 groups
                int k  = i4 >> 5;             // 0..31
                int n4 = (i4 & 31) * 4;       // 0..124
                float4 v = *(const float4*)&B[(size_t)(k0 + k) * N + bn + n4];
                lsB[(n4 + 0) * 32 + k] = f32_to_bf16(v.x);
                lsB[(n4 + 1) * 32 + k] = f32_to_bf16(v.y);
                lsB[(n4 + 2) * 32 + k] = f32_to_bf16(v.z);
                lsB[(n4 + 3) * 32 + k] = f32_to_bf16(v.w);
            }
        }
        // hint next A tile into cache (global_prefetch_b8)
        if (k0 + 32 < K) {
            if (A_BF16) __builtin_prefetch(&Ah[(size_t)(bm + srow) * K + k0 + 32], 0, 1);
            else        __builtin_prefetch(&Af[(size_t)(bm + srow) * K + k0 + 32], 0, 1);
        }
#if HAVE_ASYNC_LDS
        if (A_BF16) wait_asynccnt0();
#endif
        __syncthreads();

        // ---- load fragments per ISA 16-bit A/B layouts and issue WMMA ----
        Frag a[2], b[4];
        #pragma unroll
        for (int mi = 0; mi < 2; ++mi) {
            int arow = wm + mi * 16 + lrow;
            const unsigned int* ap = (const unsigned int*)&lsA[arow * 32 + lhi * 8];
            #pragma unroll
            for (int p = 0; p < 8; ++p) a[mi].u[p] = ap[p < 4 ? p : p + 4];
        }
        #pragma unroll
        for (int ni = 0; ni < 4; ++ni) {
            int brow = wn + ni * 16 + lrow;
            const unsigned int* bp = (const unsigned int*)&lsB[brow * 32 + lhi * 8];
            #pragma unroll
            for (int p = 0; p < 8; ++p) b[ni].u[p] = bp[p < 4 ? p : p + 4];
        }
        #pragma unroll
        for (int mi = 0; mi < 2; ++mi)
            #pragma unroll
            for (int ni = 0; ni < 4; ++ni)
                acc[mi][ni] = __builtin_amdgcn_wmma_f32_16x16x32_bf16(
                    false, a[mi].v, false, b[ni].v,
                    (short)0, acc[mi][ni], false, false);

        __syncthreads();
    }

    // ---- epilogue: bias (+relu), write f32 and/or bf16 ----
    #pragma unroll
    for (int mi = 0; mi < 2; ++mi) {
        #pragma unroll
        for (int ni = 0; ni < 4; ++ni) {
            int gc = bn + wn + ni * 16 + lrow;
            float bsv = bias ? bias[gc] : 0.0f;
            #pragma unroll
            for (int r = 0; r < 8; ++r) {
                int gr = bm + wm + mi * 16 + lhi * 8 + r;
                float v = acc[mi][ni][r] + bsv;
                if (RELU) v = fmaxf(v, 0.0f);
                if (outF) outF[(size_t)gr * N + gc] = v;
                if (outH) outH[(size_t)gr * N + gc] = f32_to_bf16(v);
            }
        }
    }
}

// ---------------------------------------------------------------------------
// Small helper kernels
// ---------------------------------------------------------------------------
__global__ void softmax64_kernel(const float* __restrict__ x, float* __restrict__ y) {
    __shared__ float buf[64];
    int t = threadIdx.x;  // 64 threads
    float v = x[t];
    buf[t] = v;
    __syncthreads();
    float m = buf[0];
    for (int i = 1; i < 64; ++i) m = fmaxf(m, buf[i]);
    __syncthreads();
    float e = __expf(v - m);
    buf[t] = e;
    __syncthreads();
    float s = 0.0f;
    for (int i = 0; i < 64; ++i) s += buf[i];
    y[t] = e / s;
}

// seg[p] = mean over rows [s,e) of video_v (T x 256)
__global__ __launch_bounds__(256)
void segment_mean_kernel(const float* __restrict__ video_v, const int* __restrict__ ts,
                         float* __restrict__ seg) {
    int p = blockIdx.x, c = threadIdx.x;
    int s = ts[2 * p];
    int e = ts[2 * p + 1];
    if (e < s + 1) e = s + 1;
    float acc = 0.0f;
    for (int r = s; r < e; ++r) acc += video_v[(size_t)r * 256 + c];
    seg[p * 256 + c] = acc / (float)(e - s);
}

// video_seg[c] = sum_p score[p]*seg[p][c]; text_seg[c] = sum_p score[p]*text_out[p][c]
__global__ __launch_bounds__(256)
void combine_seg_kernel(const float* __restrict__ score, const float* __restrict__ seg,
                        const float* __restrict__ text_out,
                        float* __restrict__ vseg, float* __restrict__ tseg) {
    int c = threadIdx.x;
    float a = 0.0f, b = 0.0f;
    for (int p = 0; p < 64; ++p) {
        float w = score[p];
        a += w * seg[p * 256 + c];
        b += w * text_out[p * 256 + c];
    }
    vseg[c] = a;
    tseg[c] = b;
}

// textcat (2176 x 768): rows [0,64)=para, 64=question, [65,2113)=a_texts, rest 0
__global__ __launch_bounds__(256)
void assemble_textcat_kernel(const float* __restrict__ para, const float* __restrict__ question,
                             const float* __restrict__ a_texts, float* __restrict__ out) {
    int i = blockIdx.x * 256 + threadIdx.x;
    if (i >= 2176 * 768) return;
    int r = i / 768, c = i % 768;
    float v;
    if (r < 64)        v = para[r * 768 + c];
    else if (r == 64)  v = question[c];
    else if (r < 2113) v = a_texts[(size_t)(r - 65) * 768 + c];
    else               v = 0.0f;
    out[i] = v;
}

// X (2048 x 1024) = [video_seg | text_seg | q + a_text_mlp | btn]
__global__ __launch_bounds__(256)
void assemble_X_kernel(const float* __restrict__ vseg, const float* __restrict__ tseg,
                       const float* __restrict__ text_out, const float* __restrict__ btn,
                       float* __restrict__ X) {
    int i = blockIdx.x * 256 + threadIdx.x;  // 2048*1024 total
    int r = i >> 10, c = i & 1023;
    float v;
    if (c < 256)      v = vseg[c];
    else if (c < 512) v = tseg[c - 256];
    else if (c < 768) v = text_out[64 * 256 + (c - 512)]
                        + text_out[(size_t)(65 + r) * 256 + (c - 512)];
    else              v = btn[(size_t)r * 256 + (c - 768)];
    X[i] = v;
}

// ---------------------------------------------------------------------------
// GRU scan (sequential, 128 steps). gi was hoisted into a big WMMA GEMM (GI).
// One persistent workgroup; state, gates, hidden in LDS; ds_add_f32 for
// the logits reduction.
// ---------------------------------------------------------------------------
__global__ __launch_bounds__(256)
void gru_scan_kernel(const float* __restrict__ inputs_pre, const float* __restrict__ GI,
                     const float* __restrict__ w_hh, const float* __restrict__ b_hh,
                     const float* __restrict__ wj1, const float* __restrict__ bj1,
                     const float* __restrict__ wj2, const float* __restrict__ bj2,
                     const float* __restrict__ state0, float* __restrict__ out) {
    __shared__ float state[512];
    __shared__ float gh[1536];
    __shared__ float hnew[16 * 512];
    __shared__ float logits[16];
    __shared__ int amax;

    int t = threadIdx.x;
    for (int c = t; c < 512; c += 256) state[c] = state0[c];
    __syncthreads();

    for (int s = 0; s < 128; ++s) {
        // gh = state @ w_hh.T + b_hh  (all 16 rows identical -> one vector)
        for (int j = t; j < 1536; j += 256) {
            const float* wr = &w_hh[(size_t)j * 512];
            float acc = b_hh[j];
            for (int k = 0; k < 512; ++k) acc += state[k] * wr[k];
            gh[j] = acc;
        }
        if (t < 16) logits[t] = bj2[0];
        __syncthreads();

        // gates + h_new
        for (int i = t; i < 16 * 512; i += 256) {
            int a = i >> 9, c = i & 511;
            size_t row = (size_t)(s * 16 + a);
            float ir  = GI[row * 1536 + c];
            float iz  = GI[row * 1536 + 512 + c];
            float inn = GI[row * 1536 + 1024 + c];
            float r = 1.0f / (1.0f + __expf(-(ir + gh[c])));
            float z = 1.0f / (1.0f + __expf(-(iz + gh[512 + c])));
            float n = tanhf(inn + r * gh[1024 + c]);
            hnew[i] = (1.0f - z) * n + z * state[c];
        }
        __syncthreads();

        // logits = relu(h_new @ wj1 + bj1) @ wj2 + bj2 (ds_add_f32 reduce)
        for (int i = t; i < 16 * 512; i += 256) {
            int a = i >> 9, c = i & 511;
            float acc = bj1[c];
            const float* hr = &hnew[a * 512];
            for (int k = 0; k < 512; ++k) acc += hr[k] * wj1[(size_t)k * 512 + c];
            atomicAdd(&logits[a], fmaxf(acc, 0.0f) * wj2[c]);
        }
        __syncthreads();

        if (t < 16) out[s * 16 + t] = logits[t];
        if (t == 0) {
            int best = 0;
            float bv = logits[0];
            for (int a = 1; a < 16; ++a)
                if (logits[a] > bv) { bv = logits[a]; best = a; }
            amax = best;
        }
        __syncthreads();

        for (int c = t; c < 512; c += 256)
            state[c] = inputs_pre[(size_t)(s * 16 + amax) * 512 + c];
        __syncthreads();
    }
}

// ---------------------------------------------------------------------------
// Launcher
// ---------------------------------------------------------------------------
extern "C" void kernel_launch(void* const* d_in, const int* in_sizes, int n_in,
                              void* d_out, int out_size, void* d_ws, size_t ws_size,
                              hipStream_t stream) {
    const float* video       = (const float*)d_in[0];
    const float* para        = (const float*)d_in[1];
    const float* question    = (const float*)d_in[2];
    const float* paras_score = (const float*)d_in[3];
    const float* a_texts     = (const float*)d_in[4];
    const float* a_buttons   = (const float*)d_in[5];
    const int*   timestamps  = (const int*)d_in[6];
    const float* wv1 = (const float*)d_in[7];
    const float* bv1 = (const float*)d_in[8];
    const float* wv2 = (const float*)d_in[9];
    const float* bv2 = (const float*)d_in[10];
    const float* wt1 = (const float*)d_in[11];
    const float* bt1 = (const float*)d_in[12];
    const float* wt2 = (const float*)d_in[13];
    const float* bt2 = (const float*)d_in[14];
    const float* wp1 = (const float*)d_in[15];
    const float* bp1 = (const float*)d_in[16];
    const float* wp2 = (const float*)d_in[17];
    const float* bp2 = (const float*)d_in[18];
    const float* w_ih = (const float*)d_in[19];
    const float* b_ih = (const float*)d_in[20];
    const float* w_hh = (const float*)d_in[21];
    const float* b_hh = (const float*)d_in[22];
    const float* wj1 = (const float*)d_in[23];
    const float* bj1 = (const float*)d_in[24];
    const float* wj2 = (const float*)d_in[25];
    const float* bj2 = (const float*)d_in[26];
    const float* state0 = (const float*)d_in[27];

    unsigned char* ws = (unsigned char*)d_ws;
    size_t off = 0;
    auto alloc = [&](size_t bytes) -> void* {
        void* p = ws + off;
        off = (off + bytes + 255) & ~(size_t)255;
        return p;
    };

    unsigned short* Hv       = (unsigned short*)alloc((size_t)16384 * 1024 * 2);
    float*          video_v  = (float*)alloc((size_t)16384 * 256 * 4);
    float*          textcat  = (float*)alloc((size_t)2176 * 768 * 4);
    unsigned short* text_h   = (unsigned short*)alloc((size_t)2176 * 768 * 2);
    float*          text_out = (float*)alloc((size_t)2176 * 256 * 4);
    unsigned short* Hb       = (unsigned short*)alloc((size_t)2048 * 1024 * 2);
    float*          btn      = (float*)alloc((size_t)2048 * 256 * 4);
    float*          X        = (float*)alloc((size_t)2048 * 1024 * 4);
    unsigned short* Hp       = (unsigned short*)alloc((size_t)2048 * 1024 * 2);
    float*          ipre_f   = (float*)alloc((size_t)2048 * 512 * 4);
    unsigned short* ipre_h   = (unsigned short*)alloc((size_t)2048 * 512 * 2);
    float*          GI       = (float*)alloc((size_t)2048 * 1536 * 4);
    float*          seg      = (float*)alloc((size_t)64 * 256 * 4);
    float*          score    = (float*)alloc(64 * 4);
    float*          vseg     = (float*)alloc(256 * 4);
    float*          tseg     = (float*)alloc(256 * 4);

    // --- video MLP: 16384x1024 -> relu -> 16384x256 ---
    gemm_wmma<false, false, true ><<<dim3(1024 / 128, 16384 / 128), 256, 0, stream>>>(
        video, wv1, bv1, nullptr, Hv, 16384, 1024, 1024);
    gemm_wmma<true,  false, false><<<dim3(256 / 128, 16384 / 128), 256, 0, stream>>>(
        Hv, wv2, bv2, video_v, nullptr, 16384, 256, 1024);

    // --- text MLP on [para; question; a_texts] ---
    assemble_textcat_kernel<<<(2176 * 768 + 255) / 256, 256, 0, stream>>>(
        para, question, a_texts, textcat);
    gemm_wmma<false, false, true ><<<dim3(768 / 128, 2176 / 128), 256, 0, stream>>>(
        textcat, wt1, bt1, nullptr, text_h, 2176, 768, 768);
    gemm_wmma<true,  false, false><<<dim3(256 / 128, 2176 / 128), 256, 0, stream>>>(
        text_h, wt2, bt2, text_out, nullptr, 2176, 256, 768);

    // --- softmax + segment means + weighted combines ---
    softmax64_kernel<<<1, 64, 0, stream>>>(paras_score, score);
    segment_mean_kernel<<<64, 256, 0, stream>>>(video_v, timestamps, seg);
    combine_seg_kernel<<<1, 256, 0, stream>>>(score, seg, text_out, vseg, tseg);

    // --- buttons MLP: 2048x1024 -> relu -> 2048x256 ---
    gemm_wmma<false, false, true ><<<dim3(1024 / 128, 2048 / 128), 256, 0, stream>>>(
        a_buttons, wv1, bv1, nullptr, Hb, 2048, 1024, 1024);
    gemm_wmma<true,  false, false><<<dim3(256 / 128, 2048 / 128), 256, 0, stream>>>(
        Hb, wv2, bv2, btn, nullptr, 2048, 256, 1024);

    // --- concat + pre-GRU MLP: 2048x1024 -> relu -> 2048x512 ---
    assemble_X_kernel<<<(2048 * 1024) / 256, 256, 0, stream>>>(
        vseg, tseg, text_out, btn, X);
    gemm_wmma<false, false, true ><<<dim3(1024 / 128, 2048 / 128), 256, 0, stream>>>(
        X, wp1, bp1, nullptr, Hp, 2048, 1024, 1024);
    gemm_wmma<true,  false, false><<<dim3(512 / 128, 2048 / 128), 256, 0, stream>>>(
        Hp, wp2, bp2, ipre_f, ipre_h, 2048, 512, 1024);

    // --- hoisted GI = inputs_pre @ w_ih.T + b_ih (state-independent) ---
    gemm_wmma<true, true, false><<<dim3(1536 / 128, 2048 / 128), 256, 0, stream>>>(
        ipre_h, w_ih, b_ih, GI, nullptr, 2048, 1536, 512);

    // --- sequential GRU scan (128 steps) ---
    gru_scan_kernel<<<1, 256, 0, stream>>>(
        ipre_f, GI, w_hh, b_hh, wj1, bj1, wj2, bj2, state0, (float*)d_out);
}
